// NeuroGNN_Encoder_24773371363443
// MI455X (gfx1250) — matrix operand order repaired and checked by reading, not compiled, not run on hardware
//
#include <hip/hip_runtime.h>

typedef __attribute__((ext_vector_type(16))) _Float16 v16h;
typedef __attribute__((ext_vector_type(8)))  _Float16 v8h;
typedef __attribute__((ext_vector_type(8)))  float    v8f;

// ---------------------------------------------------------------------------
// WMMA GEMM:  C[M,N] = act( A[M,K](f16) * Bt[N,K](f16)^T + bias[N] )
// One wave computes a 32x64 tile: 2 M-subtiles x 4 N-subtiles = 8 independent
// v_wmma_f32_16x16x32_f16 per K-step (back-to-back XDL issue, no acc RAW).
// A layout (contiguous K order, matching B): lane<16 -> K in {k0..k0+15},
// lane>=16 -> +16 offset.  B transposed ([N,K]) so lanes read contiguous K.
// K must be a multiple of 32; N a multiple of 16 (all call sites use mult of
// 64); M/N edges handled by clamped loads + guarded stores (EXEC stays all-1).
// ---------------------------------------------------------------------------
__global__ __launch_bounds__(128) void k_gemm_f16(
    const _Float16* __restrict__ A, const _Float16* __restrict__ Bt,
    const float* __restrict__ bias, float* __restrict__ C,
    int M, int N, int K, int act)
{
  const int lane = threadIdx.x & 31;
  const int wave = threadIdx.x >> 5;
  const int ncol0 = (blockIdx.x * 4 + wave) * 64;   // 64 columns per wave
  const int tm = blockIdx.y;                        // 32 rows per wave
  if (ncol0 >= N) return;

  const int half = lane >> 4;
  const int l16  = lane & 15;
  const int m0 = tm * 32 + l16;
  const int m1 = m0 + 16;
  const int m0c = (m0 < M) ? m0 : (M - 1);
  const int m1c = (m1 < M) ? m1 : (M - 1);

  const _Float16* a0 = A + (long)m0c * K + (half ? 16 : 0);
  const _Float16* a1 = A + (long)m1c * K + (half ? 16 : 0);
  const _Float16* bp[4];
#pragma unroll
  for (int j = 0; j < 4; ++j) {
    int nr = ncol0 + j * 16 + l16;
    if (nr > N - 1) nr = N - 1;
    bp[j] = Bt + (long)nr * K + (half ? 16 : 0);
  }

  v8f zero = {0.f,0.f,0.f,0.f,0.f,0.f,0.f,0.f};
  v8f acc[8];
#pragma unroll
  for (int i = 0; i < 8; ++i) acc[i] = zero;

  for (int k0 = 0; k0 < K; k0 += 32) {
    __builtin_prefetch(a0 + k0 + 128, 0, 1);   // global_prefetch_b8 (next chunks)
    v8h a0lo = *(const v8h*)(a0 + k0);
    v8h a0hi = *(const v8h*)(a0 + k0 + 8);
    v8h a1lo = *(const v8h*)(a1 + k0);
    v8h a1hi = *(const v8h*)(a1 + k0 + 8);
    v16h av0 = __builtin_shufflevector(a0lo, a0hi,
                  0,1,2,3,4,5,6,7,8,9,10,11,12,13,14,15);
    v16h av1 = __builtin_shufflevector(a1lo, a1hi,
                  0,1,2,3,4,5,6,7,8,9,10,11,12,13,14,15);
#pragma unroll
    for (int j = 0; j < 4; ++j) {
      v16h bv = *(const v16h*)(bp[j] + k0);
      acc[j]     = __builtin_amdgcn_wmma_f32_16x16x32_f16(
                       false, av0, false, bv, (short)0, acc[j], false, false);
      acc[4 + j] = __builtin_amdgcn_wmma_f32_16x16x32_f16(
                       false, av1, false, bv, (short)0, acc[4 + j], false, false);
    }
  }

#pragma unroll
  for (int j = 0; j < 4; ++j) {
    int nc = ncol0 + j * 16 + l16;
    if (nc < N) {
      float bv = bias ? bias[nc] : 0.f;
#pragma unroll
      for (int r = 0; r < 8; ++r) {
        int mm = tm * 32 + (half ? (r + 8) : r);
        if (mm < M) {
          float v = acc[j][r] + bv;
          if (act) v = fmaxf(v, 0.f);
          C[(long)mm * N + nc] = v;
        }
        int mm2 = mm + 16;
        if (mm2 < M) {
          float v = acc[4 + j][r] + bv;
          if (act) v = fmaxf(v, 0.f);
          C[(long)mm2 * N + nc] = v;
        }
      }
    }
  }
}

// ---------------- conversions / data movement ------------------------------
__global__ void k_cvt(const float* __restrict__ s, _Float16* __restrict__ d,
                      int R, int K, int Kp)
{
  long i = (long)blockIdx.x * blockDim.x + threadIdx.x;
  if (i >= (long)R * Kp) return;
  int k = (int)(i % Kp); long r = i / Kp;
  d[i] = (k < K) ? (_Float16)s[r * K + k] : (_Float16)0.f;
}

__global__ void k_cvt_T(const float* __restrict__ s, _Float16* __restrict__ d,
                        int K, int N, int Kp)
{
  long i = (long)blockIdx.x * blockDim.x + threadIdx.x;
  if (i >= (long)N * Kp) return;
  int k = (int)(i % Kp); long n = i / Kp;
  d[i] = (k < K) ? (_Float16)s[(long)k * N + n] : (_Float16)0.f;
}

// x [B,S,N,D] f32 -> xr [B*N, S, Dp] f16 (transpose + pad)
__global__ void k_prep_x(const float* __restrict__ x, _Float16* __restrict__ xr,
                         int B, int S, int N, int D, int Dp)
{
  long i = (long)blockIdx.x * blockDim.x + threadIdx.x;
  long total = (long)B * N * S * Dp;
  if (i >= total) return;
  int d = (int)(i % Dp); long t = i / Dp;
  int s = (int)(t % S);  long bn = t / S;
  int n = (int)(bn % N); int b = (int)(bn / N);
  xr[i] = (d < D) ? (_Float16)x[(((long)b * S + s) * N + n) * D + d]
                  : (_Float16)0.f;
}

__global__ void k_zero32(float* p, long n)
{ long i = (long)blockIdx.x * blockDim.x + threadIdx.x; if (i < n) p[i] = 0.f; }
__global__ void k_zero16(_Float16* p, long n)
{ long i = (long)blockIdx.x * blockDim.x + threadIdx.x; if (i < n) p[i] = (_Float16)0.f; }

__global__ void k_copy_cols(const float* __restrict__ src, float* __restrict__ dst,
                            int rows, int cols, int dstLD, int dstOff)
{
  long i = (long)blockIdx.x * blockDim.x + threadIdx.x;
  if (i >= (long)rows * cols) return;
  long r = i / cols; int c = (int)(i % cols);
  dst[r * dstLD + dstOff + c] = src[i];
}

// ---------------- layernorm (one wave per row), optional residual+relu -----
__global__ __launch_bounds__(32) void k_layernorm(
    const float* __restrict__ x, const float* __restrict__ res,
    const float* __restrict__ g, const float* __restrict__ b,
    float* __restrict__ out, int L, int relu)
{
  int row = blockIdx.x, lane = threadIdx.x;
  const float* xr = x + (long)row * L;
  const float* rr = res ? res + (long)row * L : nullptr;
  float s = 0.f;
  for (int i = lane; i < L; i += 32) s += xr[i] + (rr ? rr[i] : 0.f);
  for (int o = 16; o > 0; o >>= 1) s += __shfl_xor(s, o, 32);
  float mean = s / L;
  float vs = 0.f;
  for (int i = lane; i < L; i += 32) {
    float v = xr[i] + (rr ? rr[i] : 0.f) - mean; vs += v * v;
  }
  for (int o = 16; o > 0; o >>= 1) vs += __shfl_xor(vs, o, 32);
  float rstd = rsqrtf(vs / L + 1e-5f);
  float* orow = out + (long)row * L;
  for (int i = lane; i < L; i += 32) {
    float v = (xr[i] + (rr ? rr[i] : 0.f) - mean) * rstd * g[i] + b[i];
    if (relu) v = fmaxf(v, 0.f);
    orow[i] = v;
  }
}

// ---------------- GRU gate pointwise (PyTorch order r,z,n) -----------------
__global__ void k_gru_gate(const float* __restrict__ gi, int t,
                           const float* __restrict__ gh,
                           float* __restrict__ h, _Float16* __restrict__ h16,
                           int BN, int S, int H)
{
  long idx = (long)blockIdx.x * blockDim.x + threadIdx.x;
  if (idx >= (long)BN * H) return;
  int j = (int)(idx % H); long bn = idx / H;
  const float* gir = gi + ((long)bn * S + t) * 3 * H;
  const float* ghr = gh + (long)bn * 3 * H;
  float r = 1.f / (1.f + expf(-(gir[j]         + ghr[j])));
  float z = 1.f / (1.f + expf(-(gir[H + j]     + ghr[H + j])));
  float n = tanhf(gir[2 * H + j] + r * ghr[2 * H + j]);
  float hv = (1.f - z) * n + z * h[idx];
  h[idx] = hv;
  h16[idx] = (_Float16)hv;
}

// ---------------- attention: scores, softmax + mean over (b,h) -------------
__global__ void k_scores(const float* __restrict__ qk, float* __restrict__ sc)
{
  int idx = blockIdx.x * blockDim.x + threadIdx.x; // [B,NH,N,N]
  if (idx >= 32 * 8 * 25 * 25) return;
  int j = idx % 25, i = (idx / 25) % 25, h = (idx / 625) % 8, b = idx / 5000;
  const float* q = qk + (long)(b * 25 + i) * 2048 + h * 128;
  const float* k = qk + (long)(b * 25 + j) * 2048 + 1024 + h * 128;
  float s = 0.f;
  for (int d = 0; d < 128; ++d) s += q[d] * k[d];
  sc[idx] = s * 0.0883883476f; // 1/sqrt(128)
}

__global__ __launch_bounds__(32) void k_soft_mean(
    const float* __restrict__ sc, float* __restrict__ attn)
{
  int row = blockIdx.x, lane = threadIdx.x; // row = (b*8+h)*25+i
  float v = (lane < 25) ? sc[(long)row * 25 + lane] : -3.0e38f;
  float m = v;
  for (int o = 16; o > 0; o >>= 1) m = fmaxf(m, __shfl_xor(m, o, 32));
  float e = (lane < 25) ? expf(v - m) : 0.f;
  float s = e;
  for (int o = 16; o > 0; o >>= 1) s += __shfl_xor(s, o, 32);
  if (lane < 25) atomicAdd(&attn[(row % 25) * 25 + lane], (e / s) * (1.f / 256.f));
}

__global__ __launch_bounds__(32) void k_row_invnorm(
    const float* __restrict__ te, float* __restrict__ inv, int L)
{
  int row = blockIdx.x, lane = threadIdx.x;
  float s = 0.f;
  for (int i = lane; i < L; i += 32) { float v = te[(long)row * L + i]; s += v * v; }
  for (int o = 16; o > 0; o >>= 1) s += __shfl_xor(s, o, 32);
  if (lane == 0) inv[row] = 1.f / fmaxf(sqrtf(s), 1e-8f);
}

__global__ void k_embed_att(const float* __restrict__ te,
                            const float* __restrict__ inv, float* __restrict__ ea)
{
  int idx = blockIdx.x * blockDim.x + threadIdx.x;
  if (idx >= 625) return;
  int i = idx / 25, j = idx % 25;
  float s = 0.f;
  for (int d = 0; d < 512; ++d) s += te[i * 512 + d] * te[j * 512 + d];
  ea[idx] = s * inv[i] * inv[j];
}

// blend + case-amplification mask -> adj (single block, global max)
__global__ __launch_bounds__(1024) void k_adj(
    const float* __restrict__ attn, const float* __restrict__ dist,
    const float* __restrict__ ea, const float* __restrict__ alpha_p,
    float* __restrict__ adj)
{
  __shared__ float wm[32];
  int t = threadIdx.x;
  float alpha = alpha_p[0];
  float att = 0.f, m = 0.f;
  if (t < 625) {
    att = (alpha * dist[t] + (1.f - alpha) * ea[t]) * attn[t];
    m = powf(fmaxf(att, 0.f), 2.5f);
  }
  float w = m;
  for (int o = 16; o > 0; o >>= 1) w = fmaxf(w, __shfl_xor(w, o, 32));
  if ((t & 31) == 0) wm[t >> 5] = w;
  __syncthreads();
  if (t < 32) {
    float v = wm[t];
    for (int o = 16; o > 0; o >>= 1) v = fmaxf(v, __shfl_xor(v, o, 32));
    wm[t] = v;
  }
  __syncthreads();
  float mmax = wm[0];
  if (t < 625) adj[t] = (m >= 0.08f * mmax) ? att : 0.f;
}

// out[b,i,f] = sum_j adj[j,i] * Y[b,j,f] + bias[f]   (GCN source->target)
__global__ void k_adjmul(const float* __restrict__ Y, const float* __restrict__ adj,
                         const float* __restrict__ bias, float* __restrict__ out,
                         int F, int relu)
{
  long idx = (long)blockIdx.x * blockDim.x + threadIdx.x;
  if (idx >= (long)32 * 25 * F) return;
  int f = (int)(idx % F); int i = (int)((idx / F) % 25); int b = (int)(idx / ((long)F * 25));
  float s = bias ? bias[f] : 0.f;
  for (int j = 0; j < 25; ++j)
    s += adj[j * 25 + i] * Y[((long)b * 25 + j) * F + f];
  if (relu) s = fmaxf(s, 0.f);
  out[idx] = s;
}

// Xcat[r, 0:512]=Xt[r], Xcat[r, 512:1024]=te[r%25]
__global__ void k_build_xcat(const float* __restrict__ Xt, const float* __restrict__ te,
                             float* __restrict__ Xcat)
{
  long idx = (long)blockIdx.x * blockDim.x + threadIdx.x;
  if (idx >= (long)800 * 1024) return;
  int j = (int)(idx % 1024); long r = idx / 1024;
  Xcat[idx] = (j < 512) ? Xt[r * 512 + j] : te[(r % 25) * 512 + (j - 512)];
}

// Xfull16[r, 0:1024]=Xc16[r], Xfull16[r, 1024:1280]=(f16)Xg[r]
__global__ void k_build_xfull(const _Float16* __restrict__ Xc16,
                              const float* __restrict__ Xg,
                              _Float16* __restrict__ Xfull)
{
  long idx = (long)blockIdx.x * blockDim.x + threadIdx.x;
  if (idx >= (long)800 * 1280) return;
  int j = (int)(idx % 1280); long r = idx / 1280;
  Xfull[idx] = (j < 1024) ? Xc16[r * 1024 + j] : (_Float16)Xg[r * 256 + (j - 1024)];
}

// ---------------------------------------------------------------------------
extern "C" void kernel_launch(void* const* d_in, const int* in_sizes, int n_in,
                              void* d_out, int out_size, void* d_ws, size_t ws_size,
                              hipStream_t stream)
{
  (void)in_sizes; (void)n_in; (void)out_size; (void)ws_size;
  const float* x       = (const float*)d_in[0];
  const float* s_w1    = (const float*)d_in[1];
  const float* s_b1    = (const float*)d_in[2];
  const float* s_w2    = (const float*)d_in[3];
  const float* s_b2    = (const float*)d_in[4];
  const float* ln2_g   = (const float*)d_in[5];
  const float* ln2_b   = (const float*)d_in[6];
  const float* wih_f   = (const float*)d_in[7];
  const float* whh_f   = (const float*)d_in[8];
  const float* bih_f   = (const float*)d_in[9];
  const float* bhh_f   = (const float*)d_in[10];
  const float* wih_b   = (const float*)d_in[11];
  const float* whh_b   = (const float*)d_in[12];
  const float* bih_b   = (const float*)d_in[13];
  const float* bhh_b   = (const float*)d_in[14];
  const float* ln_g    = (const float*)d_in[15];
  const float* ln_b    = (const float*)d_in[16];
  const float* attn_w  = (const float*)d_in[17];
  const float* attn_b  = (const float*)d_in[18];
  const float* fcta_w  = (const float*)d_in[19];
  const float* fcta_b  = (const float*)d_in[20];
  const float* ln3_g   = (const float*)d_in[21];
  const float* ln3_b   = (const float*)d_in[22];
  const float* sem_e   = (const float*)d_in[23];
  const float* sem_w   = (const float*)d_in[24];
  const float* sem_b   = (const float*)d_in[25];
  const float* lnsem_g = (const float*)d_in[26];
  const float* lnsem_b = (const float*)d_in[27];
  const float* alpha   = (const float*)d_in[28];
  const float* dist    = (const float*)d_in[29];
  const float* g0w     = (const float*)d_in[30];
  const float* g0b     = (const float*)d_in[31];
  const float* g1w     = (const float*)d_in[32];
  const float* g1b     = (const float*)d_in[33];
  const float* g2w     = (const float*)d_in[34];
  const float* g2b     = (const float*)d_in[35];
  const float* gln1_g  = (const float*)d_in[36];
  const float* gln1_b  = (const float*)d_in[37];
  const float* gln2_g  = (const float*)d_in[38];
  const float* gln2_b  = (const float*)d_in[39];
  const float* eln_g   = (const float*)d_in[40];
  const float* eln_b   = (const float*)d_in[41];
  const float* fc_w    = (const float*)d_in[42];
  const float* fc_b    = (const float*)d_in[43];
  float* out = (float*)d_out;

  const int B = 32, S = 60, N = 25, D = 100, Dp = 128;
  const int BN = B * N;            // 800
  const int BNS = BN * S;          // 48000
  const int H = 512, G3 = 3 * H;   // 1536
  const int H2 = 1024;

  char* base = (char*)d_ws; size_t off = 0;
  auto alloc = [&](size_t bytes) -> void* {
    void* p = base + off; off += (bytes + 255) & ~(size_t)255; return p;
  };

  _Float16* xr16  = (_Float16*)alloc((size_t)BNS * Dp * 2);
  float*    h1    = (float*)   alloc((size_t)BNS * 256 * 4);
  _Float16* h116  = (_Float16*)alloc((size_t)BNS * 256 * 2);
  float*    h2    = (float*)   alloc((size_t)BNS * 512 * 4);
  _Float16* h16   = (_Float16*)alloc((size_t)BNS * 512 * 2);
  float*    gi    = (float*)   alloc((size_t)BNS * G3 * 4);
  float*    gh    = (float*)   alloc((size_t)BN * G3 * 4);
  float*    hbuf  = (float*)   alloc((size_t)BN * H * 4);
  _Float16* hbuf16= (_Float16*)alloc((size_t)BN * H * 2);
  float*    hn    = (float*)   alloc((size_t)BN * H2 * 4);
  _Float16* hn16  = (_Float16*)alloc((size_t)BN * H2 * 2);
  float*    qk    = (float*)   alloc((size_t)BN * 2048 * 4);
  float*    sc    = (float*)   alloc((size_t)160000 * 4);
  float*    attn  = (float*)   alloc(625 * 4);
  float*    te    = (float*)   alloc((size_t)25 * 512 * 4);
  float*    invn  = (float*)   alloc(32 * 4);
  float*    ea    = (float*)   alloc(625 * 4);
  float*    adj   = (float*)   alloc(625 * 4);
  _Float16* sem16 = (_Float16*)alloc((size_t)25 * 768 * 2);
  float*    Xt    = (float*)   alloc((size_t)BN * 512 * 4);
  float*    Xcat  = (float*)   alloc((size_t)BN * 1024 * 4);
  _Float16* Xc16  = (_Float16*)alloc((size_t)BN * 1024 * 2);
  float*    Y     = (float*)   alloc((size_t)BN * 256 * 4);
  float*    Z     = (float*)   alloc((size_t)BN * 256 * 4);
  float*    XgA   = (float*)   alloc((size_t)BN * 256 * 4);
  float*    XgB   = (float*)   alloc((size_t)BN * 256 * 4);
  _Float16* Xg16  = (_Float16*)alloc((size_t)BN * 256 * 2);
  _Float16* Xfull = (_Float16*)alloc((size_t)BN * 1280 * 2);
  _Float16* w1t   = (_Float16*)alloc((size_t)256 * 128 * 2);
  _Float16* w2t   = (_Float16*)alloc((size_t)512 * 256 * 2);
  _Float16* wihf16= (_Float16*)alloc((size_t)G3 * 512 * 2);
  _Float16* whhf16= (_Float16*)alloc((size_t)G3 * 512 * 2);
  _Float16* wihb16= (_Float16*)alloc((size_t)G3 * 512 * 2);
  _Float16* whhb16= (_Float16*)alloc((size_t)G3 * 512 * 2);
  _Float16* aw16  = (_Float16*)alloc((size_t)2048 * 1024 * 2);
  _Float16* ftat  = (_Float16*)alloc((size_t)512 * 1024 * 2);
  _Float16* semt  = (_Float16*)alloc((size_t)512 * 768 * 2);
  _Float16* g0t   = (_Float16*)alloc((size_t)256 * 1024 * 2);
  _Float16* g1t   = (_Float16*)alloc((size_t)256 * 256 * 2);
  _Float16* g2t   = (_Float16*)alloc((size_t)256 * 256 * 2);
  _Float16* fct   = (_Float16*)alloc((size_t)512 * 1280 * 2);

  auto blocks = [](long n) { return dim3((unsigned)((n + 255) / 256)); };
  auto gemm = [&](const _Float16* A, const _Float16* Bt, const float* bias,
                  float* C, int M, int Nn, int K, int act) {
    dim3 grid((unsigned)((Nn / 64 + 3) / 4), (unsigned)((M + 31) / 32));
    k_gemm_f16<<<grid, 128, 0, stream>>>(A, Bt, bias, C, M, Nn, K, act);
  };

  // ---- weight conversions to f16 (Bt = [N,K] layouts) ----
  k_cvt_T<<<blocks((long)256 * 128), 256, 0, stream>>>(s_w1, w1t, 100, 256, 128);
  k_cvt_T<<<blocks((long)512 * 256), 256, 0, stream>>>(s_w2, w2t, 256, 512, 256);
  k_cvt  <<<blocks((long)G3 * 512), 256, 0, stream>>>(wih_f, wihf16, G3, 512, 512);
  k_cvt  <<<blocks((long)G3 * 512), 256, 0, stream>>>(whh_f, whhf16, G3, 512, 512);
  k_cvt  <<<blocks((long)G3 * 512), 256, 0, stream>>>(wih_b, wihb16, G3, 512, 512);
  k_cvt  <<<blocks((long)G3 * 512), 256, 0, stream>>>(whh_b, whhb16, G3, 512, 512);
  k_cvt  <<<blocks((long)2048 * 1024), 256, 0, stream>>>(attn_w, aw16, 2048, 1024, 1024);
  k_cvt_T<<<blocks((long)512 * 1024), 256, 0, stream>>>(fcta_w, ftat, 1024, 512, 1024);
  k_cvt_T<<<blocks((long)512 * 768), 256, 0, stream>>>(sem_w, semt, 768, 512, 768);
  k_cvt  <<<blocks((long)25 * 768), 256, 0, stream>>>(sem_e, sem16, 25, 768, 768);
  k_cvt_T<<<blocks((long)256 * 1024), 256, 0, stream>>>(g0w, g0t, 1024, 256, 1024);
  k_cvt_T<<<blocks((long)256 * 256), 256, 0, stream>>>(g1w, g1t, 256, 256, 256);
  k_cvt_T<<<blocks((long)256 * 256), 256, 0, stream>>>(g2w, g2t, 256, 256, 256);
  k_cvt_T<<<blocks((long)512 * 1280), 256, 0, stream>>>(fc_w, fct, 1280, 512, 1280);

  // ---- seq1 MLP + LN ----
  k_prep_x<<<blocks((long)BNS * Dp), 256, 0, stream>>>(x, xr16, B, S, N, D, Dp);
  gemm(xr16, w1t, s_b1, h1, BNS, 256, 128, 1);
  k_cvt<<<blocks((long)BNS * 256), 256, 0, stream>>>(h1, h116, BNS, 256, 256);
  gemm(h116, w2t, s_b2, h2, BNS, 512, 256, 1);
  k_layernorm<<<BNS, 32, 0, stream>>>(h2, nullptr, ln2_g, ln2_b, h2, 512, 0);
  k_cvt<<<blocks((long)BNS * 512), 256, 0, stream>>>(h2, h16, BNS, 512, 512);

  // ---- bidirectional GRU (gi buffer reused across directions) ----
  for (int dir = 0; dir < 2; ++dir) {
    const _Float16* wih = dir ? wihb16 : wihf16;
    const _Float16* whh = dir ? whhb16 : whhf16;
    const float* bih = dir ? bih_b : bih_f;
    const float* bhh = dir ? bhh_b : bhh_f;
    gemm(h16, wih, bih, gi, BNS, G3, 512, 0);
    k_zero32<<<blocks((long)BN * H), 256, 0, stream>>>(hbuf, (long)BN * H);
    k_zero16<<<blocks((long)BN * H), 256, 0, stream>>>(hbuf16, (long)BN * H);
    for (int step = 0; step < S; ++step) {
      int t = dir ? (S - 1 - step) : step;
      gemm(hbuf16, whh, bhh, gh, BN, G3, 512, 0);
      k_gru_gate<<<blocks((long)BN * H), 256, 0, stream>>>(
          gi, t, gh, hbuf, hbuf16, BN, S, H);
    }
    k_copy_cols<<<blocks((long)BN * H), 256, 0, stream>>>(
        hbuf, hn, BN, H, H2, dir ? H : 0);
  }
  k_layernorm<<<BN, 32, 0, stream>>>(hn, nullptr, ln_g, ln_b, hn, H2, 0);
  k_cvt<<<blocks((long)BN * H2), 256, 0, stream>>>(hn, hn16, BN, H2, H2);

  // ---- multihead attention weights (q,k only; v unused) ----
  gemm(hn16, aw16, attn_b, qk, BN, 2048, H2, 0);
  k_scores<<<blocks(160000), 256, 0, stream>>>(qk, sc);
  k_zero32<<<blocks(625), 256, 0, stream>>>(attn, 625);
  k_soft_mean<<<6400, 32, 0, stream>>>(sc, attn);

  // ---- node features ----
  gemm(hn16, ftat, fcta_b, Xt, BN, 512, H2, 1);            // relu(fc_ta)
  k_layernorm<<<BN, 32, 0, stream>>>(Xt, nullptr, ln3_g, ln3_b, Xt, 512, 0);
  gemm(sem16, semt, sem_b, te, 25, 512, 768, 0);           // semantic proj
  k_layernorm<<<25, 32, 0, stream>>>(te, nullptr, lnsem_g, lnsem_b, te, 512, 0);
  k_build_xcat<<<blocks((long)BN * 1024), 256, 0, stream>>>(Xt, te, Xcat);

  // ---- adjacency ----
  k_row_invnorm<<<25, 32, 0, stream>>>(te, invn, 512);
  k_embed_att<<<blocks(625), 256, 0, stream>>>(te, invn, ea);
  k_adj<<<1, 1024, 0, stream>>>(attn, dist, ea, alpha, adj);

  // ---- GCN block ----
  k_cvt<<<blocks((long)BN * 1024), 256, 0, stream>>>(Xcat, Xc16, BN, 1024, 1024);
  gemm(Xc16, g0t, nullptr, Y, BN, 256, 1024, 0);
  k_adjmul<<<blocks((long)BN * 256), 256, 0, stream>>>(Y, adj, g0b, XgA, 256, 1);
  k_cvt<<<blocks((long)BN * 256), 256, 0, stream>>>(XgA, Xg16, BN, 256, 256);
  gemm(Xg16, g1t, nullptr, Y, BN, 256, 256, 0);
  k_adjmul<<<blocks((long)BN * 256), 256, 0, stream>>>(Y, adj, g1b, Z, 256, 0);
  k_layernorm<<<BN, 32, 0, stream>>>(Z, XgA, gln1_g, gln1_b, XgB, 256, 1);
  k_cvt<<<blocks((long)BN * 256), 256, 0, stream>>>(XgB, Xg16, BN, 256, 256);
  gemm(Xg16, g2t, nullptr, Y, BN, 256, 256, 0);
  k_adjmul<<<blocks((long)BN * 256), 256, 0, stream>>>(Y, adj, g2b, Z, 256, 0);
  k_layernorm<<<BN, 32, 0, stream>>>(Z, XgB, gln2_g, gln2_b, XgA, 256, 1);

  // ---- encoder head ----
  k_layernorm<<<BN, 32, 0, stream>>>(XgA, nullptr, eln_g, eln_b, XgB, 256, 0);
  k_build_xfull<<<blocks((long)BN * 1280), 256, 0, stream>>>(Xc16, XgB, Xfull);
  gemm(Xfull, fct, fc_b, out, BN, 512, 1280, 1);
}
